// EncoderBlock_37718402793908
// MI455X (gfx1250) — compile-verified, matrix operand order checked by reading, not run on hardware
//
#include <hip/hip_runtime.h>
#include <hip/hip_bf16.h>

#define B_   8
#define S_   1024
#define D_   1024
#define H_   16
#define DK_  64
#define DFF_ 4096

typedef __attribute__((ext_vector_type(16))) __bf16 v16bf;
typedef __attribute__((ext_vector_type(8)))  __bf16 v8bf;
typedef __attribute__((ext_vector_type(8)))  float  v8f;
typedef __attribute__((ext_vector_type(4)))  unsigned int v4u;
typedef __attribute__((ext_vector_type(8)))  int v8i;
typedef __attribute__((ext_vector_type(4)))  int v4i;

__device__ __forceinline__ v8f wmma_bf16(v16bf a, v16bf b, v8f c) {
  // D = A(16x32 bf16) x B(32x16 bf16) + C(16x16 f32)
  return __builtin_amdgcn_wmma_f32_16x16x32_bf16(
      /*neg_a=*/false, a, /*neg_b=*/false, b,
      /*c_mod=*/(short)0, c, /*reuse_a=*/false, /*reuse_b=*/false);
}

// Fragment loader for 16x32 A (row-major rows) or 32x16 B taken from an
// N-major (transposed) source: lane&15 selects the row/column, (lane>>4)
// selects the K-half per the ISA 16-bit layout (K 0..7 / 8..15 then +16).
__device__ __forceinline__ v16bf load_frag(const __bf16* __restrict__ base,
                                           int stride) {
  const int lane = threadIdx.x & 31;
  const int r    = lane & 15;
  const int kb   = (lane >> 4) << 3;  // 0 or 8
  const v8bf lo = *(const v8bf*)(base + (size_t)r * stride + kb);
  const v8bf hi = *(const v8bf*)(base + (size_t)r * stride + kb + 16);
  v16bf a;
#pragma unroll
  for (int j = 0; j < 8; ++j) { a[j] = lo[j]; a[j + 8] = hi[j]; }
  return a;
}

// Same fragment pattern but reading a TDM-staged LDS tile laid out as
// 64 columns x 32 K-elements, column-contiguous (64 B per column).
__device__ __forceinline__ v16bf load_frag_lds(const __bf16* tile, int nt) {
  const int lane = threadIdx.x & 31;
  const int r    = lane & 15;
  const int kb   = (lane >> 4) << 3;
  const __bf16* col = tile + (nt * 16 + r) * 32 + kb;
  const v8bf lo = *(const v8bf*)col;
  const v8bf hi = *(const v8bf*)(col + 16);
  v16bf a;
#pragma unroll
  for (int j = 0; j < 8; ++j) { a[j] = lo[j]; a[j + 8] = hi[j]; }
  return a;
}

// Tensor Data Mover: DMA a 2D bf16 tile [tile_d1 rows x tile_d0 elems]
// from global (row stride = tensor_d0 elements) into LDS at lds_off.
// D# packing per CDNA5 ISA.
// This toolchain exposes the 6-arg builtin:
//   (v4u g0, v8i g1, v4i g2, v4i g3, v8i g4, i32 cpol)
__device__ __forceinline__ void tdm_load_tile_bf16(const __bf16* gptr,
                                                   unsigned lds_off,
                                                   unsigned tensor_d0,
                                                   unsigned tensor_d1,
                                                   unsigned tile_d0,
                                                   unsigned tile_d1) {
  const unsigned long long ga = (unsigned long long)(uintptr_t)gptr;
  v4u g0;
  g0[0] = 1u;                                   // count=1, user mode
  g0[1] = lds_off;                              // lds_addr (bytes)
  g0[2] = (unsigned)(ga & 0xffffffffu);         // global_addr[31:0]
  g0[3] = (unsigned)((ga >> 32) & 0x01ffffffu)  // global_addr[56:32]
          | (2u << 30);                         // type=2 ("image")
  v8i g1;
  g1[0] = (int)(1u << 16);                      // data_size=1 -> 2 bytes
  g1[1] = (int)((tensor_d0 & 0xffffu) << 16);   // tensor_dim0[15:0]
  g1[2] = (int)(((tensor_d0 >> 16) & 0xffffu)   // tensor_dim0[31:16]
          | ((tensor_d1 & 0xffffu) << 16));     // tensor_dim1[15:0]
  g1[3] = (int)(((tensor_d1 >> 16) & 0xffffu)   // tensor_dim1[31:16]
          | (tile_d0 << 16));                   // tile_dim0
  g1[4] = (int)(tile_d1 & 0xffffu);             // tile_dim1 (tile_dim2=0)
  g1[5] = (int)tensor_d0;                       // tensor_dim0_stride[31:0]
  g1[6] = 0;                                    // stride hi, dim1_stride lo
  g1[7] = 0;
  const v4i z4 = {0, 0, 0, 0};
  const v8i z8 = {0, 0, 0, 0, 0, 0, 0, 0};
  __builtin_amdgcn_tensor_load_to_lds(g0, g1, z4, z4, z8, 0);
}

// ---------------- weight transpose + cast: W[K,N] f32 -> Wt[N,K] bf16 -------
__global__ __launch_bounds__(256) void transpose_cast_kernel(
    const float* __restrict__ w, __bf16* __restrict__ wt, int K, int N) {
  const size_t idx = (size_t)blockIdx.x * 256 + threadIdx.x;
  if (idx >= (size_t)K * N) return;
  const int n  = (int)(idx / K);
  const int kk = (int)(idx % K);
  wt[idx] = (__bf16)w[(size_t)kk * N + n];
}

// ---------------- row LayerNorm, f32 in -> bf16 out -------------------------
__global__ __launch_bounds__(256) void ln_bf16_kernel(
    const float* __restrict__ x, const float* __restrict__ g,
    const float* __restrict__ bta, __bf16* __restrict__ out) {
  const int row = blockIdx.x;
  const int tid = threadIdx.x;
  const float4 v = ((const float4*)(x + (size_t)row * D_))[tid];
  float s = v.x + v.y + v.z + v.w;
  float q = v.x * v.x + v.y * v.y + v.z * v.z + v.w * v.w;
#pragma unroll
  for (int off = 16; off > 0; off >>= 1) {
    s += __shfl_xor(s, off, 32);
    q += __shfl_xor(q, off, 32);
  }
  __shared__ float rs[8], rq[8];
  const int wv = tid >> 5, ln = tid & 31;
  if (ln == 0) { rs[wv] = s; rq[wv] = q; }
  __syncthreads();
  s = 0.0f; q = 0.0f;
#pragma unroll
  for (int i = 0; i < 8; ++i) { s += rs[i]; q += rq[i]; }
  const float mu   = s * (1.0f / D_);
  const float var  = q * (1.0f / D_) - mu * mu;
  const float rstd = rsqrtf(var + 1e-5f);
  const float4 gg = ((const float4*)g)[tid];
  const float4 bb = ((const float4*)bta)[tid];
  __bf16* o = out + (size_t)row * D_ + tid * 4;
  o[0] = (__bf16)((v.x - mu) * rstd * gg.x + bb.x);
  o[1] = (__bf16)((v.y - mu) * rstd * gg.y + bb.y);
  o[2] = (__bf16)((v.z - mu) * rstd * gg.z + bb.z);
  o[3] = (__bf16)((v.w - mu) * rstd * gg.w + bb.w);
}

// ---------------- WMMA GEMM with TDM-staged B tiles -------------------------
// out = A[M,K]bf16 @ Wt[N,K]bf16^T + bias.  Block = 4 waves, 64x64 tile.
// Per K-step the 4 KB Wt tile is DMA'd once per block by the Tensor Data
// Mover (double buffered), instead of 4x redundant per-wave global loads.
constexpr int MODE_BF16      = 0;  // bf16 store, row-major
constexpr int MODE_BF16_VT   = 1;  // bf16 store as V^T [B,H,DK,S]
constexpr int MODE_BF16_RELU = 2;  // relu then bf16 store
constexpr int MODE_F32_RESID = 3;  // f32 store, + residual

template <int MODE>
__global__ __launch_bounds__(128) void gemm_wmma_kernel(
    const __bf16* __restrict__ A, const __bf16* __restrict__ Bt,
    const float* __restrict__ bias, const float* __restrict__ resid,
    void* __restrict__ out, int M, int N, int K) {
  __shared__ __bf16 sB[2][64 * 32];  // two 4 KB tiles (64 cols x 32 K)
  const int wave = threadIdx.x >> 5;
  const int lane = threadIdx.x & 31;
  const int m0 = blockIdx.y * 64 + wave * 16;
  const int n0 = blockIdx.x * 64;

  const unsigned lds0 = (unsigned)(uintptr_t)&sB[0][0];
  const unsigned lds1 = (unsigned)(uintptr_t)&sB[1][0];

  // prologue: DMA first tile (wave 0 only; TDM ignores EXEC)
  if (threadIdx.x < 32) {
    tdm_load_tile_bf16(Bt + (size_t)n0 * K, lds0, (unsigned)K, (unsigned)N,
                       32u, 64u);
    __builtin_amdgcn_s_wait_tensorcnt(0);
  }
  __syncthreads();

  v8f c[4] = {};
  const __bf16* arow = A + (size_t)m0 * K;
  int buf = 0;
  for (int k0 = 0; k0 < K; k0 += 32) {
    // kick off DMA of the next tile into the other buffer
    if (k0 + 32 < K && threadIdx.x < 32) {
      tdm_load_tile_bf16(Bt + (size_t)n0 * K + (k0 + 32),
                         buf ? lds0 : lds1, (unsigned)K, (unsigned)N,
                         32u, 64u);
    }
    const v16bf a = load_frag(arow + k0, K);
#pragma unroll
    for (int nt = 0; nt < 4; ++nt) {
      const v16bf b = load_frag_lds(&sB[buf][0], nt);
      c[nt] = wmma_bf16(a, b, c[nt]);
    }
    if (threadIdx.x < 32) __builtin_amdgcn_s_wait_tensorcnt(0);
    __syncthreads();
    buf ^= 1;
  }

  const int nloc = lane & 15;
  const int rb   = (lane >> 4) << 3;
#pragma unroll
  for (int nt = 0; nt < 4; ++nt) {
    const int n = n0 + nt * 16 + nloc;
    const float bi = bias[n];
#pragma unroll
    for (int r = 0; r < 8; ++r) {
      const int m = m0 + rb + r;
      float val = c[nt][r] + bi;
      if (MODE == MODE_BF16_RELU) val = fmaxf(val, 0.0f);
      if (MODE == MODE_F32_RESID) {
        ((float*)out)[(size_t)m * N + n] = resid[(size_t)m * N + n] + val;
      } else if (MODE == MODE_BF16_VT) {
        const int bb = m >> 10, ss = m & (S_ - 1);
        const int hh = n >> 6,  dd = n & (DK_ - 1);
        ((__bf16*)out)[((size_t)(bb * H_ + hh) * DK_ + dd) * S_ + ss] =
            (__bf16)val;
      } else {
        ((__bf16*)out)[(size_t)m * N + n] = (__bf16)val;
      }
    }
  }
}

// ---------------- attention: one wave per (b, h, 16-query tile) -------------
__global__ __launch_bounds__(32) void attn_kernel(
    const __bf16* __restrict__ q, const __bf16* __restrict__ k,
    const __bf16* __restrict__ vt, const int* __restrict__ mask,
    __bf16* __restrict__ ctx) {
  __shared__ float sS[16 * S_];  // 64 KB score/probability tile
  const int qt = blockIdx.x, hh = blockIdx.y, bb = blockIdx.z;
  const int lane = threadIdx.x;
  const int q0 = qt * 16;
  const float scale = 0.125f;  // 1/sqrt(64)

  const __bf16* qbase = q + ((size_t)(bb * S_ + q0)) * D_ + hh * DK_;
  const v16bf aq0 = load_frag(qbase, D_);
  const v16bf aq1 = load_frag(qbase + 32, D_);

  const int nloc = lane & 15;
  const int rb   = (lane >> 4) << 3;

  // scores = scale * Q @ K^T  -> LDS
  for (int kt = 0; kt < S_ / 16; ++kt) {
    const __bf16* kbase = k + ((size_t)(bb * S_ + kt * 16)) * D_ + hh * DK_;
    v8f acc = {};
    acc = wmma_bf16(aq0, load_frag(kbase, D_), acc);
    acc = wmma_bf16(aq1, load_frag(kbase + 32, D_), acc);
#pragma unroll
    for (int r = 0; r < 8; ++r)
      sS[(rb + r) * S_ + kt * 16 + nloc] = acc[r] * scale;
  }
  __syncthreads();

  // masked softmax: lane&15 = row, half-wave splits the key range
  const int row = lane & 15;
  const int hb  = (lane >> 4) * (S_ / 2);
  const int* mrow = mask + bb * S_;
  float* srow = sS + row * S_;
  float mx = -1e30f;
  for (int i = 0; i < S_ / 2; ++i) {
    float sv = srow[hb + i];
    if (mrow[hb + i] == 0) sv = -1e9f;
    srow[hb + i] = sv;
    mx = fmaxf(mx, sv);
  }
  mx = fmaxf(mx, __shfl_xor(mx, 16, 32));
  float sum = 0.0f;
  for (int i = 0; i < S_ / 2; ++i) {
    const float e = __expf(srow[hb + i] - mx);
    srow[hb + i] = e;
    sum += e;
  }
  sum += __shfl_xor(sum, 16, 32);
  const float inv = 1.0f / sum;
  for (int i = 0; i < S_ / 2; ++i) srow[hb + i] *= inv;
  __syncthreads();

  // ctx = P @ V  (V stored transposed: vt[B,H,DK,S] -> contiguous B frags)
  v8f c[4] = {};
  const __bf16* vbase0 = vt + (size_t)(bb * H_ + hh) * DK_ * S_;
  for (int kc = 0; kc < S_; kc += 32) {
    const int pr = lane & 15;
    const int kb = (lane >> 4) << 3;
    v16bf ap;
#pragma unroll
    for (int j = 0; j < 8; ++j) {
      ap[j]     = (__bf16)sS[pr * S_ + kc + kb + j];
      ap[j + 8] = (__bf16)sS[pr * S_ + kc + kb + 16 + j];
    }
#pragma unroll
    for (int nt = 0; nt < 4; ++nt) {
      const v16bf bf = load_frag(vbase0 + (size_t)(nt * 16) * S_ + kc, S_);
      c[nt] = wmma_bf16(ap, bf, c[nt]);
    }
  }
#pragma unroll
  for (int nt = 0; nt < 4; ++nt) {
#pragma unroll
    for (int r = 0; r < 8; ++r) {
      const int m = q0 + rb + r;
      const int d = nt * 16 + nloc;
      ctx[((size_t)(bb * S_ + m)) * D_ + hh * DK_ + d] = (__bf16)c[nt][r];
    }
  }
}

// ---------------------------------------------------------------------------
extern "C" void kernel_launch(void* const* d_in, const int* in_sizes, int n_in,
                              void* d_out, int out_size, void* d_ws,
                              size_t ws_size, hipStream_t stream) {
  (void)in_sizes; (void)n_in; (void)out_size; (void)ws_size;
  const float* x   = (const float*)d_in[0];
  const int*   msk = (const int*)d_in[1];
  const float* wq  = (const float*)d_in[2];
  const float* bq  = (const float*)d_in[3];
  const float* wk  = (const float*)d_in[4];
  const float* bk  = (const float*)d_in[5];
  const float* wv  = (const float*)d_in[6];
  const float* bv  = (const float*)d_in[7];
  const float* wo  = (const float*)d_in[8];
  const float* bo  = (const float*)d_in[9];
  const float* w1  = (const float*)d_in[10];
  const float* b1  = (const float*)d_in[11];
  const float* w2  = (const float*)d_in[12];
  const float* b2  = (const float*)d_in[13];
  const float* g1  = (const float*)d_in[14];
  const float* be1 = (const float*)d_in[15];
  const float* g2  = (const float*)d_in[16];
  const float* be2 = (const float*)d_in[17];
  float* out = (float*)d_out;

  char* ws = (char*)d_ws;
  const size_t MB = 1024 * 1024;
  __bf16* wqt    = (__bf16*)(ws + 0 * MB);    // 2 MB each
  __bf16* wkt    = (__bf16*)(ws + 2 * MB);
  __bf16* wvt    = (__bf16*)(ws + 4 * MB);
  __bf16* wot    = (__bf16*)(ws + 6 * MB);
  __bf16* w1t    = (__bf16*)(ws + 8 * MB);    // 8 MB
  __bf16* w2t    = (__bf16*)(ws + 16 * MB);   // 8 MB
  __bf16* h_bf   = (__bf16*)(ws + 24 * MB);   // 16 MB
  __bf16* q_bf   = (__bf16*)(ws + 40 * MB);   // 16 MB
  __bf16* k_bf   = (__bf16*)(ws + 56 * MB);   // 16 MB
  __bf16* vt_bf  = (__bf16*)(ws + 72 * MB);   // 16 MB
  __bf16* ctx_bf = (__bf16*)(ws + 88 * MB);   // 16 MB
  float*  x1     = (float*)(ws + 104 * MB);   // 32 MB
  __bf16* h2_bf  = (__bf16*)(ws + 136 * MB);  // 16 MB
  __bf16* ff1_bf = (__bf16*)(ws + 152 * MB);  // 64 MB

  const int MT = B_ * S_;  // 8192 tokens

  // weights -> bf16, N-major
  {
    size_t e = (size_t)D_ * D_;
    dim3 g((unsigned)((e + 255) / 256));
    transpose_cast_kernel<<<g, 256, 0, stream>>>(wq, wqt, D_, D_);
    transpose_cast_kernel<<<g, 256, 0, stream>>>(wk, wkt, D_, D_);
    transpose_cast_kernel<<<g, 256, 0, stream>>>(wv, wvt, D_, D_);
    transpose_cast_kernel<<<g, 256, 0, stream>>>(wo, wot, D_, D_);
    size_t ef = (size_t)D_ * DFF_;
    dim3 gf((unsigned)((ef + 255) / 256));
    transpose_cast_kernel<<<gf, 256, 0, stream>>>(w1, w1t, D_, DFF_);
    transpose_cast_kernel<<<gf, 256, 0, stream>>>(w2, w2t, DFF_, D_);
  }

  // LN1
  ln_bf16_kernel<<<MT, 256, 0, stream>>>(x, g1, be1, h_bf);

  // Q, K, V projections
  dim3 gm(D_ / 64, MT / 64);  // (16,128)
  gemm_wmma_kernel<MODE_BF16><<<gm, 128, 0, stream>>>(
      h_bf, wqt, bq, nullptr, q_bf, MT, D_, D_);
  gemm_wmma_kernel<MODE_BF16><<<gm, 128, 0, stream>>>(
      h_bf, wkt, bk, nullptr, k_bf, MT, D_, D_);
  gemm_wmma_kernel<MODE_BF16_VT><<<gm, 128, 0, stream>>>(
      h_bf, wvt, bv, nullptr, vt_bf, MT, D_, D_);

  // attention
  dim3 ga(S_ / 16, H_, B_);
  attn_kernel<<<ga, 32, 0, stream>>>(q_bf, k_bf, vt_bf, msk, ctx_bf);

  // output projection + residual (f32)
  gemm_wmma_kernel<MODE_F32_RESID><<<gm, 128, 0, stream>>>(
      ctx_bf, wot, bo, x, x1, MT, D_, D_);

  // LN2
  ln_bf16_kernel<<<MT, 256, 0, stream>>>(x1, g2, be2, h2_bf);

  // FFN
  dim3 gff(DFF_ / 64, MT / 64);  // (64,128)
  gemm_wmma_kernel<MODE_BF16_RELU><<<gff, 128, 0, stream>>>(
      h2_bf, w1t, b1, nullptr, ff1_bf, MT, DFF_, D_);
  gemm_wmma_kernel<MODE_F32_RESID><<<gm, 128, 0, stream>>>(
      ff1_bf, w2t, b2, x1, out, MT, D_, DFF_);
}
